// MoE_5617817224061
// MI455X (gfx1250) — matrix operand order repaired and checked by reading, not compiled, not run on hardware
//
#include <hip/hip_runtime.h>

typedef __bf16 bf16;
typedef __attribute__((ext_vector_type(16))) __bf16 v16bf;
typedef __attribute__((ext_vector_type(8)))  __bf16 v8bf;
typedef __attribute__((ext_vector_type(8)))  float  v8f;

static constexpr int Bz = 4, Sz = 1024, Dz = 512, Fz = 1024, Ez = 8, Kz = 2;
static constexpr int Tz = Bz * Sz;                 // 4096 tokens
static constexpr float LB_W = 0.01f, Z_W = 0.001f;

// ---------------- conversion / transpose ----------------
__global__ void k_f2bf(const float* __restrict__ in, bf16* __restrict__ out, int n) {
    int i = blockIdx.x * blockDim.x + threadIdx.x;
    if (i < n) out[i] = (bf16)in[i];
}

// in: [E][R][C] fp32  ->  out: [E][C][R] bf16   (linear over output)
__global__ void k_transpose_bf16(const float* __restrict__ in, bf16* __restrict__ out,
                                 int R, int C) {
    int i = blockIdx.x * blockDim.x + threadIdx.x;
    int total = Ez * R * C;
    if (i >= total) return;
    int r = i % R;
    int tmp = i / R;
    int c = tmp % C;
    int e = tmp / C;
    out[i] = (bf16)in[((size_t)e * R + r) * C + c];
}

// ---------------- router: one wave32 per token ----------------
__global__ void k_router(const float* __restrict__ x, const float* __restrict__ rw,
                         int* __restrict__ topi2, float* __restrict__ topw,
                         float* __restrict__ probs, float* __restrict__ lse2) {
    int wv = threadIdx.x >> 5, lane = threadIdx.x & 31;
    int t = blockIdx.x * 8 + wv;
    float acc[Ez];
#pragma unroll
    for (int e = 0; e < Ez; ++e) acc[e] = 0.f;
    for (int d = lane; d < Dz; d += 32) {
        float xv = x[(size_t)t * Dz + d];
#pragma unroll
        for (int e = 0; e < Ez; ++e) acc[e] += xv * rw[e * Dz + d];
    }
#pragma unroll
    for (int e = 0; e < Ez; ++e) {
#pragma unroll
        for (int off = 16; off > 0; off >>= 1)
            acc[e] += __shfl_xor(acc[e], off, 32);
    }
    if (lane == 0) {
        float mx = acc[0];
#pragma unroll
        for (int e = 1; e < Ez; ++e) mx = fmaxf(mx, acc[e]);
        float p[Ez]; float s = 0.f;
#pragma unroll
        for (int e = 0; e < Ez; ++e) { p[e] = __expf(acc[e] - mx); s += p[e]; }
        float inv = 1.f / s;
#pragma unroll
        for (int e = 0; e < Ez; ++e) { p[e] *= inv; probs[t * Ez + e] = p[e]; }
        float lse = mx + __logf(s);
        lse2[t] = lse * lse;
        int i1 = 0;
#pragma unroll
        for (int e = 1; e < Ez; ++e) if (p[e] > p[i1]) i1 = e;
        int i2 = (i1 == 0) ? 1 : 0;
#pragma unroll
        for (int e = 0; e < Ez; ++e) if (e != i1 && e != i2 && p[e] > p[i2]) i2 = e;
        float wsum = p[i1] + p[i2];
        topi2[t * 2 + 0] = i1;
        topi2[t * 2 + 1] = i2;
        topw[t * 2 + 0] = p[i1] / wsum;
        topw[t * 2 + 1] = p[i2] / wsum;
    }
}

// ---------------- deterministic per-expert list build (block prefix scan) ----
__global__ void k_build_lists(const int* __restrict__ topi2, int* __restrict__ list,
                              int* __restrict__ counts) {
    const int e = blockIdx.x;
    const int tid = threadIdx.x;
    const int NT = 1024;
    __shared__ int sc[NT];
    __shared__ int running;
    if (tid == 0) running = 0;
    __syncthreads();
    for (int base = 0; base < Tz; base += NT) {
        int t = base + tid;
        int match = -1;
        int e0 = topi2[t * 2], e1 = topi2[t * 2 + 1];
        if (e0 == e) match = t * 2;
        else if (e1 == e) match = t * 2 + 1;
        int flag = (match >= 0) ? 1 : 0;
        sc[tid] = flag;
        __syncthreads();
        for (int off = 1; off < NT; off <<= 1) {
            int v = (tid >= off) ? sc[tid - off] : 0;
            __syncthreads();
            sc[tid] += v;
            __syncthreads();
        }
        int incl = sc[tid];
        int roff = running;
        __syncthreads();
        if (match >= 0) list[e * Tz + roff + incl - 1] = match;
        if (tid == NT - 1) running = roff + incl;
        __syncthreads();
    }
    if (tid == NT - 1) counts[e] = running;
}

__global__ void k_zero(float* __restrict__ out, int n) {
    int i = blockIdx.x * blockDim.x + threadIdx.x;
    if (i < n) out[i] = 0.f;
}

// ---------------- gate/up grouped GEMM + SwiGLU -> hidden (bf16) -------------
// grid: (Tz/32 row tiles, Fz/128, Ez). block 256 = 8 waves.
// Each wave: 32(M) x 16(N) tile = two A fragments sharing each B fragment.
__global__ void k_gateup(const bf16* __restrict__ xb, const bf16* __restrict__ gT,
                         const bf16* __restrict__ uT, const int* __restrict__ list,
                         const int* __restrict__ counts, bf16* __restrict__ hidden) {
    const int e = blockIdx.z;
    const int cnt = counts[e];
    const int row0 = blockIdx.x * 32;
    if (row0 >= cnt) return;                 // uniform across block
    const int wv = threadIdx.x >> 5;
    const int lane = threadIdx.x & 31;
    const int n0 = blockIdx.y * 128 + wv * 16;
    __shared__ int ent[32];
    if (threadIdx.x < 32) {
        int r = row0 + threadIdx.x;
        ent[threadIdx.x] = list[e * Tz + ((r < cnt) ? r : row0)];
    }
    __syncthreads();
    const int l15 = lane & 15;
    const int hi = lane >> 4;                // 0: K low half, 1: K high half
    const int tok0 = ent[l15] >> 1;
    const int tok1 = ent[16 + l15] >> 1;
    const bf16* arow0 = xb + (size_t)tok0 * Dz + hi * 16;
    const bf16* arow1 = xb + (size_t)tok1 * Dz + hi * 16;
    const bf16* bg = gT + ((size_t)e * Fz + (n0 + l15)) * Dz + hi * 16;
    const bf16* bu = uT + ((size_t)e * Fz + (n0 + l15)) * Dz + hi * 16;
    v8f cg0 = {}; v8f cu0 = {}; v8f cg1 = {}; v8f cu1 = {};
    for (int k0 = 0; k0 < Dz; k0 += 32) {
        v8bf a0lo = *(const v8bf*)(arow0 + k0);
        v8bf a0hi = *(const v8bf*)(arow0 + k0 + 8);
        v8bf a1lo = *(const v8bf*)(arow1 + k0);
        v8bf a1hi = *(const v8bf*)(arow1 + k0 + 8);
        v16bf a0, a1;
#pragma unroll
        for (int i = 0; i < 8; ++i) {
            a0[i] = a0lo[i]; a0[8 + i] = a0hi[i];
            a1[i] = a1lo[i]; a1[8 + i] = a1hi[i];
        }
        v16bf bgv = *(const v16bf*)(bg + k0);
        v16bf buv = *(const v16bf*)(bu + k0);
        cg0 = __builtin_amdgcn_wmma_f32_16x16x32_bf16(false, a0, false, bgv, (short)0, cg0, false, false);
        cg1 = __builtin_amdgcn_wmma_f32_16x16x32_bf16(false, a1, false, bgv, (short)0, cg1, false, false);
        cu0 = __builtin_amdgcn_wmma_f32_16x16x32_bf16(false, a0, false, buv, (short)0, cu0, false, false);
        cu1 = __builtin_amdgcn_wmma_f32_16x16x32_bf16(false, a1, false, buv, (short)0, cu1, false, false);
    }
    const int ncol = n0 + l15;
    const int mbase = hi * 8;
#pragma unroll
    for (int r = 0; r < 8; ++r) {
        int m = mbase + r;
        if (row0 + m < cnt) {
            float g = cg0[r], u = cu0[r];
            float h = g * (1.f / (1.f + __expf(-g))) * u;   // silu(g)*u
            hidden[(size_t)ent[m] * Fz + ncol] = (bf16)h;
        }
        int m2 = 16 + m;
        if (row0 + m2 < cnt) {
            float g = cg1[r], u = cu1[r];
            float h = g * (1.f / (1.f + __expf(-g))) * u;
            hidden[(size_t)ent[m2] * Fz + ncol] = (bf16)h;
        }
    }
}

// ---------------- down grouped GEMM, weighted scatter-add --------------------
// grid: (Tz/32, Dz/128, Ez). Each wave: 32(M) x 16(N) tile.
__global__ void k_down(const bf16* __restrict__ hidden, const bf16* __restrict__ dT,
                       const int* __restrict__ list, const int* __restrict__ counts,
                       const float* __restrict__ topw, float* __restrict__ out) {
    const int e = blockIdx.z;
    const int cnt = counts[e];
    const int row0 = blockIdx.x * 32;
    if (row0 >= cnt) return;
    const int wv = threadIdx.x >> 5;
    const int lane = threadIdx.x & 31;
    const int n0 = blockIdx.y * 128 + wv * 16;
    __shared__ int ent[32];
    __shared__ float wr[32];
    if (threadIdx.x < 32) {
        int r = row0 + threadIdx.x;
        int v = list[e * Tz + ((r < cnt) ? r : row0)];
        ent[threadIdx.x] = v;
        wr[threadIdx.x] = (r < cnt) ? topw[v] : 0.f;
    }
    __syncthreads();
    const int l15 = lane & 15;
    const int hi = lane >> 4;
    const bf16* arow0 = hidden + (size_t)ent[l15] * Fz + hi * 16;
    const bf16* arow1 = hidden + (size_t)ent[16 + l15] * Fz + hi * 16;
    const bf16* bd = dT + ((size_t)e * Dz + (n0 + l15)) * Fz + hi * 16;
    v8f c0 = {}; v8f c1 = {};
    for (int k0 = 0; k0 < Fz; k0 += 32) {
        v8bf a0lo = *(const v8bf*)(arow0 + k0);
        v8bf a0hi = *(const v8bf*)(arow0 + k0 + 8);
        v8bf a1lo = *(const v8bf*)(arow1 + k0);
        v8bf a1hi = *(const v8bf*)(arow1 + k0 + 8);
        v16bf a0, a1;
#pragma unroll
        for (int i = 0; i < 8; ++i) {
            a0[i] = a0lo[i]; a0[8 + i] = a0hi[i];
            a1[i] = a1lo[i]; a1[8 + i] = a1hi[i];
        }
        v16bf bv = *(const v16bf*)(bd + k0);
        c0 = __builtin_amdgcn_wmma_f32_16x16x32_bf16(false, a0, false, bv, (short)0, c0, false, false);
        c1 = __builtin_amdgcn_wmma_f32_16x16x32_bf16(false, a1, false, bv, (short)0, c1, false, false);
    }
    const int ncol = n0 + l15;
    const int mbase = hi * 8;
#pragma unroll
    for (int r = 0; r < 8; ++r) {
        int m = mbase + r;
        if (row0 + m < cnt)
            atomicAdd(&out[(size_t)(ent[m] >> 1) * Dz + ncol], wr[m] * c0[r]);
        int m2 = 16 + m;
        if (row0 + m2 < cnt)
            atomicAdd(&out[(size_t)(ent[m2] >> 1) * Dz + ncol], wr[m2] * c1[r]);
    }
}

// ---------------- aux loss (single block, fixed-order reductions) ------------
__global__ void k_finalize(const int* __restrict__ topi2, const float* __restrict__ probs,
                           const float* __restrict__ lse2, float* __restrict__ aux_out) {
    __shared__ float red[256];
    __shared__ float Pe[Ez];
    __shared__ int Ce[Ez];
    __shared__ float Zs;
    int tid = threadIdx.x;
    if (tid < Ez) Ce[tid] = 0;
    __syncthreads();
    float pe[Ez];
#pragma unroll
    for (int e = 0; e < Ez; ++e) pe[e] = 0.f;
    float z = 0.f;
    int ce[Ez];
#pragma unroll
    for (int e = 0; e < Ez; ++e) ce[e] = 0;
    for (int t = tid; t < Tz; t += 256) {
#pragma unroll
        for (int e = 0; e < Ez; ++e) pe[e] += probs[t * Ez + e];
        z += lse2[t];
        ce[topi2[t * 2]]++;
        ce[topi2[t * 2 + 1]]++;
    }
    for (int e = 0; e < Ez; ++e) {
        red[tid] = pe[e]; __syncthreads();
        for (int off = 128; off > 0; off >>= 1) {
            if (tid < off) red[tid] += red[tid + off];
            __syncthreads();
        }
        if (tid == 0) Pe[e] = red[0];
        __syncthreads();
    }
    red[tid] = z; __syncthreads();
    for (int off = 128; off > 0; off >>= 1) {
        if (tid < off) red[tid] += red[tid + off];
        __syncthreads();
    }
    if (tid == 0) Zs = red[0];
#pragma unroll
    for (int e = 0; e < Ez; ++e) if (ce[e]) atomicAdd(&Ce[e], ce[e]);
    __syncthreads();
    if (tid == 0) {
        float lb = 0.f;
        for (int e = 0; e < Ez; ++e)
            lb += ((float)Ce[e] / (float)(Tz * Kz)) * (Pe[e] / (float)Tz);
        lb *= (float)Ez;
        aux_out[0] = LB_W * lb + Z_W * (Zs / (float)Tz);
    }
}

// ---------------- launcher ----------------
extern "C" void kernel_launch(void* const* d_in, const int* in_sizes, int n_in,
                              void* d_out, int out_size, void* d_ws, size_t ws_size,
                              hipStream_t stream) {
    (void)in_sizes; (void)n_in; (void)out_size; (void)ws_size;
    const float* x  = (const float*)d_in[0];   // [T, D]
    const float* rw = (const float*)d_in[1];   // [E, D]
    const float* gw = (const float*)d_in[2];   // [E, D, F]
    const float* uw = (const float*)d_in[3];   // [E, D, F]
    const float* dw = (const float*)d_in[4];   // [E, F, D]
    float* out = (float*)d_out;                // [T*D] + aux at [T*D]

    char* ws = (char*)d_ws;
    size_t off = 0;
    auto alloc = [&](size_t bytes) -> void* {
        void* p = ws + off;
        off = (off + bytes + 255) & ~(size_t)255;
        return p;
    };
    bf16*  gT    = (bf16*)alloc(sizeof(bf16) * (size_t)Ez * Fz * Dz);   // [E][F][D]
    bf16*  uT    = (bf16*)alloc(sizeof(bf16) * (size_t)Ez * Fz * Dz);   // [E][F][D]
    bf16*  dT    = (bf16*)alloc(sizeof(bf16) * (size_t)Ez * Dz * Fz);   // [E][D][F]
    bf16*  xb    = (bf16*)alloc(sizeof(bf16) * (size_t)Tz * Dz);
    bf16*  hid   = (bf16*)alloc(sizeof(bf16) * (size_t)2 * Tz * Fz);    // keyed by t*2+s
    int*   topi2 = (int*)alloc(sizeof(int) * Tz * 2);
    float* topw  = (float*)alloc(sizeof(float) * Tz * 2);
    float* probs = (float*)alloc(sizeof(float) * Tz * Ez);
    float* lse2  = (float*)alloc(sizeof(float) * Tz);
    int*   list  = (int*)alloc(sizeof(int) * Ez * Tz);
    int*   cnts  = (int*)alloc(sizeof(int) * 256);

    const int nXD = Tz * Dz;                 // 2,097,152
    const int nW  = Ez * Dz * Fz;            // 4,194,304

    k_f2bf<<<(nXD + 255) / 256, 256, 0, stream>>>(x, xb, nXD);
    k_transpose_bf16<<<(nW + 255) / 256, 256, 0, stream>>>(gw, gT, Dz, Fz);
    k_transpose_bf16<<<(nW + 255) / 256, 256, 0, stream>>>(uw, uT, Dz, Fz);
    k_transpose_bf16<<<(nW + 255) / 256, 256, 0, stream>>>(dw, dT, Fz, Dz);

    k_router<<<Tz / 8, 256, 0, stream>>>(x, rw, topi2, topw, probs, lse2);
    k_build_lists<<<Ez, 1024, 0, stream>>>(topi2, list, cnts);
    k_zero<<<(nXD + 255) / 256, 256, 0, stream>>>(out, nXD);

    k_gateup<<<dim3(Tz / 32, Fz / 128, Ez), 256, 0, stream>>>(xb, gT, uT, list, cnts, hid);
    k_down<<<dim3(Tz / 32, Dz / 128, Ez), 256, 0, stream>>>(hid, dT, list, cnts, topw, out);

    k_finalize<<<1, 256, 0, stream>>>(topi2, probs, lse2, out + (size_t)Tz * Dz);
}
